// DeepAR_11725260718426
// MI455X (gfx1250) — matrix operand order repaired and checked by reading, not compiled, not run on hardware
//
#include <hip/hip_runtime.h>
#include <hip/hip_bf16.h>

// DeepAR single-layer LSTM + per-step linear heads, CDNA5 (gfx1250).
// Batch-parallel recurrence: 16 workgroups, each owns 16 batch rows for all
// 1024 timesteps. h/c live in LDS. Gate & head GEMMs via
// v_wmma_f32_16x16x32_bf16 with register-resident bf16 weight fragments.
// x tiles are double-buffered into LDS with global_load_async_to_lds_b128
// (ASYNCcnt) so global latency is hidden behind the recurrent GEMM.

#define SL   1024   // sequence length L
#define NB   256    // batch B
#define NIN  64     // input size
#define NK   128    // hidden size
#define NOBS 32     // observation size
#define NG   512    // 4*NK gate columns
#define NA   192    // unified A operand width: [x | h] = IN + K

typedef __attribute__((ext_vector_type(16))) __bf16 v16bf;
typedef __attribute__((ext_vector_type(8)))  float  v8f;

// ---- WMMA fragment helpers (layouts per CDNA5 ISA §7.12.2) ----

// A-matrix 16x32 bf16: lane l -> M = l&15, half = l>>4.
// VGPR v holds K pair K0(v)=kbase + (v>=4?16:0) + (v&3)*2 + half*8.
__device__ __forceinline__ v16bf load_A_bf16(const __bf16* src, int ld,
                                             int M, int half, int kbase) {
  v16bf a;
#pragma unroll
  for (int v = 0; v < 8; ++v) {
    int k0 = kbase + ((v >= 4) ? 16 : 0) + (v & 3) * 2 + half * 8;
    a[2 * v]     = src[M * ld + k0];
    a[2 * v + 1] = src[M * ld + k0 + 1];
  }
  return a;
}

// B-matrix 32x16 bf16 (B[k][n] = W[n][k], gates = a @ W^T):
// lane l -> N = l&15, half = l>>4; VGPR v holds K pair kbase + (v>=4?16:0) + (v&3)*2 + half*8.
__device__ __forceinline__ v16bf load_B_w(const float* __restrict__ W, int ld,
                                          int nbase, int lane, int kbase) {
  int n = nbase + (lane & 15);
  int kh = kbase + (lane >> 4) * 8;
  v16bf b;
#pragma unroll
  for (int v = 0; v < 8; ++v) {
    int k0 = kh + ((v >= 4) ? 16 : 0) + (v & 3) * 2;
    b[2 * v]     = (__bf16)W[n * ld + k0];
    b[2 * v + 1] = (__bf16)W[n * ld + k0 + 1];
  }
  return b;
}

__device__ __forceinline__ v8f wmma_bf16(v16bf a, v16bf b, v8f c) {
  return __builtin_amdgcn_wmma_f32_16x16x32_bf16(
      /*neg_a=*/false, a, /*neg_b=*/false, b,
      /*c_mod=*/(short)0, c, /*reuse_a=*/false, /*reuse_b=*/false);
}

__device__ __forceinline__ float fsigmoid(float x) {
  return 1.0f / (1.0f + __expf(-x));
}
__device__ __forceinline__ float ftanh(float x) {
  // tanh(x) = 1 - 2/(1+e^{2x}); saturates correctly via inf/0 at extremes
  return 1.0f - 2.0f / (1.0f + __expf(2.0f * x));
}

// Async copy of 16 bytes global -> LDS, tracked by ASYNCcnt (gfx1250).
__device__ __forceinline__ void async_g2l_b128(unsigned lds_addr, const float* gaddr) {
  asm volatile("global_load_async_to_lds_b128 %0, %1, off"
               :: "v"(lds_addr), "v"(gaddr)
               : "memory");
}
__device__ __forceinline__ void wait_asynccnt0() {
  asm volatile("s_wait_asynccnt 0x0" ::: "memory");
}

__global__ __launch_bounds__(512, 1) void deepar_lstm_kernel(
    const float* __restrict__ x,      // [L,B,IN]
    const float* __restrict__ W_ih,   // [4K, IN]
    const float* __restrict__ W_hh,   // [4K, K]
    const float* __restrict__ b_ih,   // [4K]
    const float* __restrict__ b_hh,   // [4K]
    const float* __restrict__ W_mu,   // [OBS, K]
    const float* __restrict__ b_mu,   // [OBS]
    const float* __restrict__ W_sig,  // [OBS, K]
    const float* __restrict__ b_sig,  // [OBS]
    float* __restrict__ out_mu,       // [L,B,OBS]
    float* __restrict__ out_sig)      // [L,B,OBS]
{
  __shared__ float  s_gates[16][NG];       // 32 KB pre-activation gates (i,f,g,o)
  __shared__ float  s_c[16][NK];           //  8 KB cell state (fp32)
  __shared__ __bf16 s_a[16][NA];           //  6 KB unified A operand [x | h] (bf16)
  __shared__ float  s_xstage[2][16][NIN];  //  8 KB double-buffered raw x tiles

  const int tid  = threadIdx.x;
  const int lane = tid & 31;
  const int wave = tid >> 5;          // 0..15
  const int half = lane >> 4;
  const int mrow = lane & 15;         // A row / B,C column within tile
  const int bm   = blockIdx.x * 16;   // this WG's batch rows [bm, bm+16)

  // ---- Preload loop-invariant bf16 weight fragments into VGPRs ----
  // Unified K axis: ksteps 0..1 -> W_ih (IN=64), ksteps 2..5 -> W_hh (K=128).
  // Wave w owns gate columns [32w, 32w+32): N-tiles 2w, 2w+1.
  v16bf Bg[2][6];
  float bias[2];
#pragma unroll
  for (int tt = 0; tt < 2; ++tt) {
    int nb = wave * 32 + tt * 16;
#pragma unroll
    for (int ks = 0; ks < 2; ++ks) Bg[tt][ks] = load_B_w(W_ih, NIN, nb, lane, ks * 32);
#pragma unroll
    for (int ks = 2; ks < 6; ++ks) Bg[tt][ks] = load_B_w(W_hh, NK, nb, lane, (ks - 2) * 32);
    bias[tt] = b_ih[nb + mrow] + b_hh[nb + mrow];
  }

  // Waves 0..3 additionally hold head fragments: w0/w1 -> mu tiles, w2/w3 -> sig.
  v16bf Bhead[4];
  float hbias = 0.0f;
  if (wave < 4) {
    const float* Wh = (wave < 2) ? W_mu : W_sig;
    const float* bh = (wave < 2) ? b_mu : b_sig;
    int nb = (wave & 1) * 16;
#pragma unroll
    for (int ks = 0; ks < 4; ++ks) Bhead[ks] = load_B_w(Wh, NK, nb, lane, ks * 32);
    hbias = bh[nb + mrow];
  }

  // ---- Prologue: kick off async stage of x_0 (tile is 4 KB, contiguous) ----
  if (tid < 256) {
    unsigned ldsa = (unsigned)(uintptr_t)(&s_xstage[0][0][0]) + (unsigned)tid * 16u;
    async_g2l_b128(ldsa, x + (size_t)bm * NIN + (size_t)tid * 4);
  }

  // ---- Init state (h part of s_a, and c) ----
  for (int i = tid; i < 16 * NK; i += 512) {
    s_a[i >> 7][NIN + (i & 127)] = (__bf16)0.0f;
    (&s_c[0][0])[i] = 0.0f;
  }

  // ---- Recurrence over time ----
  for (int t = 0; t < SL; ++t) {
    const int cur = t & 1;

    // x_t staging complete -> visible to all waves.
    wait_asynccnt0();
    __syncthreads();

    // Convert staged x_t (fp32) into the A operand (bf16): 1024 elems, 2/thread.
#pragma unroll
    for (int e = 0; e < 2; ++e) {
      int idx = tid * 2 + e;          // stay pairwise-contiguous per thread? keep simple
      int r = idx >> 6, c = idx & 63;
      s_a[r][c] = (__bf16)s_xstage[cur][r][c];
    }

    // Kick off async stage of x_{t+1} into the other buffer.
    if (t + 1 < SL && tid < 256) {
      unsigned ldsa = (unsigned)(uintptr_t)(&s_xstage[cur ^ 1][0][0]) + (unsigned)tid * 16u;
      async_g2l_b128(ldsa, x + ((size_t)(t + 1) * NB + bm) * NIN + (size_t)tid * 4);
    }
    __syncthreads();

    // Gate GEMM: gates[16, 32w..32w+32) = [x|h] · [W_ih|W_hh]^T + bias
    v16bf Aa[6];
#pragma unroll
    for (int ks = 0; ks < 6; ++ks) Aa[ks] = load_A_bf16(&s_a[0][0], NA, mrow, half, ks * 32);
#pragma unroll
    for (int tt = 0; tt < 2; ++tt) {
      v8f acc = {};
#pragma unroll
      for (int ks = 0; ks < 6; ++ks) acc = wmma_bf16(Aa[ks], Bg[tt][ks], acc);
      int ncol = wave * 32 + tt * 16 + mrow;
#pragma unroll
      for (int r = 0; r < 8; ++r)
        s_gates[r + half * 8][ncol] = acc[r] + bias[tt];
    }
    __syncthreads();

    // Pointwise LSTM cell: 16x128 = 2048 elements, 4 per thread.
#pragma unroll
    for (int e = 0; e < 4; ++e) {
      int idx = tid + e * 512;
      int bb = idx >> 7;      // 0..15
      int kk = idx & 127;     // 0..127
      float ig = fsigmoid(s_gates[bb][kk]);
      float fg = fsigmoid(s_gates[bb][NK + kk]);
      float gg = ftanh(s_gates[bb][2 * NK + kk]);
      float og = fsigmoid(s_gates[bb][3 * NK + kk]);
      float c  = fg * s_c[bb][kk] + ig * gg;
      float h  = og * ftanh(c);
      s_c[bb][kk] = c;
      s_a[bb][NIN + kk] = (__bf16)h;   // h feeds next step's A operand
    }
    __syncthreads();

    // Heads (waves 0..3): mu / logsigma = h·W^T + b, streamed to HBM.
    // Other waves roll into step t+1; the next iteration's first barrier keeps
    // these s_a reads ordered before any rewrite of the h region.
    if (wave < 4) {
      v16bf Ah[4];
#pragma unroll
      for (int ks = 0; ks < 4; ++ks)
        Ah[ks] = load_A_bf16(&s_a[0][NIN], NA, mrow, half, ks * 32);
      v8f acc = {};
#pragma unroll
      for (int ks = 0; ks < 4; ++ks) acc = wmma_bf16(Ah[ks], Bhead[ks], acc);
      float* outp = (wave < 2) ? out_mu : out_sig;
      int nb = (wave & 1) * 16;
#pragma unroll
      for (int r = 0; r < 8; ++r) {
        int brow = bm + r + half * 8;
        outp[((size_t)t * NB + brow) * NOBS + nb + mrow] = acc[r] + hbias;
      }
    }
  }
}

extern "C" void kernel_launch(void* const* d_in, const int* in_sizes, int n_in,
                              void* d_out, int out_size, void* d_ws, size_t ws_size,
                              hipStream_t stream) {
  (void)in_sizes; (void)n_in; (void)d_ws; (void)ws_size;
  const float* x     = (const float*)d_in[0];
  const float* W_ih  = (const float*)d_in[1];
  const float* W_hh  = (const float*)d_in[2];
  const float* b_ih  = (const float*)d_in[3];
  const float* b_hh  = (const float*)d_in[4];
  const float* W_mu  = (const float*)d_in[5];
  const float* b_mu  = (const float*)d_in[6];
  const float* W_sig = (const float*)d_in[7];
  const float* b_sig = (const float*)d_in[8];

  float* out_mu  = (float*)d_out;
  float* out_sig = (float*)d_out + (size_t)SL * NB * NOBS;
  (void)out_size;

  // 16 workgroups x 16 batch rows; 512 threads = 16 wave32 waves per WG.
  deepar_lstm_kernel<<<dim3(NB / 16), dim3(512), 0, stream>>>(
      x, W_ih, W_hh, b_ih, b_hh, W_mu, b_mu, W_sig, b_sig, out_mu, out_sig);
}
